// SAVSSBlock_42262478193184
// MI455X (gfx1250) — matrix operand order repaired and checked by reading, not compiled, not run on hardware
//
#include <hip/hip_runtime.h>

typedef __attribute__((ext_vector_type(16))) __bf16 v16bf;
typedef __attribute__((ext_vector_type(8)))  __bf16 v8bf;
typedef __attribute__((ext_vector_type(2)))  __bf16 v2bf;
typedef __attribute__((ext_vector_type(8)))  float  v8f;

static constexpr int B_ = 4, C_ = 128, H_ = 128, W_ = 128;
static constexpr int HW = H_ * W_;
static constexpr int G_ = 16, CPG = C_ / G_;          // 8 channels per group
static constexpr int CM = 16;
static constexpr size_t NEL = (size_t)B_ * C_ * HW;   // 8388608
static constexpr size_t NSM = (size_t)B_ * CM * HW;   // 1048576

#define DEVFN __device__ __forceinline__

DEVFN float sigmoidf_(float x) { return 1.f / (1.f + __expf(-x)); }
DEVFN float siluf_(float x)    { return x / (1.f + __expf(-x)); }

// ---------------------------------------------------------------------------
// fp32 -> bf16 weight pre-conversion (run once per weight matrix)
// ---------------------------------------------------------------------------
__global__ __launch_bounds__(256)
void cvt_bf16(const float* __restrict__ src, __bf16* __restrict__ dst, int n) {
    int i = blockIdx.x * 256 + threadIdx.x;
    if (i < n) dst[i] = (__bf16)src[i];
}

// ---------------------------------------------------------------------------
// Pointwise conv (1x1) as WMMA GEMM:  Y[b,o,p] = sum_c W[o,c] * X[b,c,p]
// - X tile staged in LDS K-major per pixel: B-fragments are two aligned
//   16-byte ds_load_b128 of contiguous bf16. Optional elementwise multiply
//   with X2 fused into staging (HASMUL).
// - W pre-converted to bf16: A-fragments are two aligned 16-byte global
//   loads (k contiguous in each 8-element chunk of the ISA A layout).
// - Epilogue: optional bias, silu or sigmoid, residual add.
// ---------------------------------------------------------------------------
template <int CIN, int COUT, int NTILE,
          bool HASB, bool DOSILU, bool DOSIG, bool HASRES, bool HASMUL>
__global__ __launch_bounds__(256)
void gemm_pw(const float* __restrict__ X, const float* __restrict__ X2,
             const __bf16* __restrict__ Wm,
             const float* __restrict__ bias, const float* __restrict__ res,
             float* __restrict__ Y) {
    constexpr int CINP    = (CIN < 32) ? 32 : CIN;
    constexpr int CSTR    = CINP + 8;                 // pitch: 16B multiple, bank-staggered
    constexpr int KSTEPS  = CINP / 32;
    constexpr int mTiles  = COUT / 16;
    constexpr int mClamp  = (mTiles < 8) ? mTiles : 8;
    constexpr int MT      = (mTiles + 7) / 8;
    constexpr int wavesPerM = 8 / mClamp;
    constexpr int NT      = (NTILE / 16) / wavesPerM;

    __shared__ __align__(16) __bf16 xs[NTILE][CSTR];  // [pixel][k]

    const int tid    = threadIdx.x;
    const int wv     = tid >> 5;
    const int lane   = tid & 31;
    const int laneHi = lane >> 4;
    const int lane15 = lane & 15;
    const int b      = blockIdx.y;
    const int p0     = blockIdx.x * NTILE;

    // Stage X tile fp32 -> bf16, transposed to K-major; two channels per
    // thread -> v_cvt_pk_bf16_f32 + ds_store_b32. Global reads pixel-coalesced.
    const float* xbase = X + ((size_t)b * CIN) * HW + p0;
    const float* x2base = HASMUL ? (X2 + ((size_t)b * CIN) * HW + p0) : nullptr;
    __builtin_prefetch(xbase + tid, 0, 1);            // global_prefetch_b8
    for (int i = tid; i < (CINP / 2) * NTILE; i += 256) {
        int cq = i / NTILE, n = i - cq * NTILE;
        int c  = cq * 2;
        float v0 = (c     < CIN) ? xbase[(size_t)c * HW + n]       : 0.f;
        float v1 = (c + 1 < CIN) ? xbase[(size_t)(c + 1) * HW + n] : 0.f;
        if (HASMUL) {
            if (c     < CIN) v0 *= x2base[(size_t)c * HW + n];
            if (c + 1 < CIN) v1 *= x2base[(size_t)(c + 1) * HW + n];
        }
        v2bf pk; pk[0] = (__bf16)v0; pk[1] = (__bf16)v1;
        *reinterpret_cast<v2bf*>(&xs[n][c]) = pk;
    }
    __syncthreads();

    const int mIdx0  = wv % mClamp;
    const int nStart = (wv / mClamp) * NT;

    v8f acc[MT][NT];
    #pragma unroll
    for (int a = 0; a < MT; ++a)
        #pragma unroll
        for (int t = 0; t < NT; ++t) {
            v8f z = {0.f, 0.f, 0.f, 0.f, 0.f, 0.f, 0.f, 0.f};
            acc[a][t] = z;
        }

    #pragma unroll
    for (int ks = 0; ks < KSTEPS; ++ks) {
        const int k0 = ks * 32;
        #pragma unroll
        for (int mt = 0; mt < MT; ++mt) {
            const int m = (mIdx0 + mt * 8) * 16 + lane15;
            // A fragment: elements 0..7 -> k = k0+laneHi*8+e ; 8..15 -> +16
            const __bf16* wp = Wm + (size_t)m * CIN + k0 + laneHi * 8;
            v8bf alo = *reinterpret_cast<const v8bf*>(wp);
            v8bf ahi;
            if constexpr (CIN >= 32) {
                ahi = *reinterpret_cast<const v8bf*>(wp + 16);
            } else {
                #pragma unroll
                for (int e = 0; e < 8; ++e) ahi[e] = (__bf16)0.f;
            }
            v16bf af = __builtin_shufflevector(alo, ahi,
                0, 1, 2, 3, 4, 5, 6, 7, 8, 9, 10, 11, 12, 13, 14, 15);
            #pragma unroll
            for (int nt = 0; nt < NT; ++nt) {
                const int nb = (nStart + nt) * 16 + lane15;
                const __bf16* bp = &xs[nb][k0 + laneHi * 8];
                v8bf blo = *reinterpret_cast<const v8bf*>(bp);
                v8bf bhi = *reinterpret_cast<const v8bf*>(bp + 16);
                v16bf bf = __builtin_shufflevector(blo, bhi,
                    0, 1, 2, 3, 4, 5, 6, 7, 8, 9, 10, 11, 12, 13, 14, 15);
                acc[mt][nt] = __builtin_amdgcn_wmma_f32_16x16x32_bf16(
                    false, af, false, bf, (short)0, acc[mt][nt], false, false);
            }
        }
    }

    // Epilogue per 16x16 f32 C/D layout: VGPR r -> row r (+8 for upper lanes).
    #pragma unroll
    for (int mt = 0; mt < MT; ++mt)
        #pragma unroll
        for (int nt = 0; nt < NT; ++nt)
            #pragma unroll
            for (int r = 0; r < 8; ++r) {
                int o  = (mIdx0 + mt * 8) * 16 + r + laneHi * 8;
                int pp = p0 + (nStart + nt) * 16 + lane15;
                size_t oi = ((size_t)b * COUT + o) * HW + pp;
                float v = acc[mt][nt][r];
                if (HASB)   v += bias[o];
                if (DOSILU) v  = siluf_(v);
                if (DOSIG)  v  = sigmoidf_(v);
                if (HASRES) v += res[oi];
                Y[oi] = v;
            }
}

// ---------------------------------------------------------------------------
// GroupNorm: per (b,g) mean / rstd, then apply (+optional silu, +residual)
// ---------------------------------------------------------------------------
__global__ __launch_bounds__(256)
void gn_stats(const float* __restrict__ X, float* __restrict__ stats) {
    int bg = blockIdx.x;                              // 0..B*G-1 (channel-contig)
    const float* base = X + (size_t)bg * CPG * HW;
    float s = 0.f, ss = 0.f;
    for (int i = threadIdx.x; i < CPG * HW; i += 256) {
        float v = base[i]; s += v; ss += v * v;
    }
    __shared__ float sh[512];
    sh[threadIdx.x] = s; sh[256 + threadIdx.x] = ss;
    __syncthreads();
    for (int o = 128; o > 0; o >>= 1) {
        if (threadIdx.x < o) {
            sh[threadIdx.x]       += sh[threadIdx.x + o];
            sh[256 + threadIdx.x] += sh[256 + threadIdx.x + o];
        }
        __syncthreads();
    }
    if (threadIdx.x == 0) {
        float inv = 1.f / (float)(CPG * HW);
        float m   = sh[0] * inv;
        float var = sh[256] * inv - m * m;
        stats[bg * 2]     = m;
        stats[bg * 2 + 1] = rsqrtf(var + 1e-5f);
    }
}

__global__ __launch_bounds__(256)
void gn_apply(const float* __restrict__ X, const float* __restrict__ gam,
              const float* __restrict__ bet, const float* __restrict__ stats,
              float* __restrict__ Y, int doSilu, const float* __restrict__ resid) {
    size_t i = (size_t)blockIdx.x * 256 + threadIdx.x;
    if (i >= NEL) return;
    int c  = (int)((i / HW) % C_);
    int bg = (int)(i / ((size_t)CPG * HW));
    float v = (X[i] - stats[bg * 2]) * stats[bg * 2 + 1] * gam[c] + bet[c];
    if (doSilu) v = siluf_(v);
    if (resid)  v += resid[i];
    Y[i] = v;
}

// ---------------------------------------------------------------------------
// Depthwise 3x3 (zero pad) and per-channel scale (depthwise 1x1)
// ---------------------------------------------------------------------------
template <int CN>
__global__ __launch_bounds__(256)
void dwconv3(const float* __restrict__ X, const float* __restrict__ Wd,
             float* __restrict__ Y) {
    size_t i = (size_t)blockIdx.x * 256 + threadIdx.x;
    if (i >= (size_t)B_ * CN * HW) return;
    int w = (int)(i % W_);
    int h = (int)((i / W_) % H_);
    int c = (int)((i / HW) % CN);
    const float* xp = X + (i - (size_t)h * W_ - w);
    float acc = 0.f;
    #pragma unroll
    for (int ky = -1; ky <= 1; ++ky) {
        int hh = h + ky;
        if (hh < 0 || hh >= H_) continue;
        #pragma unroll
        for (int kx = -1; kx <= 1; ++kx) {
            int ww = w + kx;
            if (ww < 0 || ww >= W_) continue;
            acc += xp[(size_t)hh * W_ + ww] * Wd[c * 9 + (ky + 1) * 3 + (kx + 1)];
        }
    }
    Y[i] = acc;
}

template <int CN>
__global__ __launch_bounds__(256)
void chan_scale(const float* __restrict__ X, const float* __restrict__ Wd,
                float* __restrict__ Y) {
    size_t i = (size_t)blockIdx.x * 256 + threadIdx.x;
    if (i >= (size_t)B_ * CN * HW) return;
    Y[i] = X[i] * Wd[(i / HW) % CN];
}

// ---------------------------------------------------------------------------
// Row scans (axis=W): lr + rl via in-wave parallel scan, 1 wave = 1 row.
// State op: s_t = a*s_{t-1} + b*x_t ; segment pair (A,S), combine
// (Ap,Sp)o(A,S) = (Ap*A, A*Sp + S).
// ---------------------------------------------------------------------------
__global__ __launch_bounds__(256)
void rowscan(const float* __restrict__ X,
             const float* __restrict__ pa, const float* __restrict__ pb,
             const float* __restrict__ pc, const float* __restrict__ pd,
             float* __restrict__ Y) {
    int row  = blockIdx.x * 8 + (threadIdx.x >> 5);   // over B*C*H rows
    int lane = threadIdx.x & 31;
    int c    = (row / H_) % C_;
    const float* x = X + (size_t)row * W_;
    float*       y = Y + (size_t)row * W_;

    float a0 = sigmoidf_(pa[c]),      b0 = pb[c],      c0 = pc[c],      d0 = pd[c];
    float a1 = sigmoidf_(pa[C_ + c]), b1 = pb[C_ + c], c1 = pc[C_ + c], d1 = pd[C_ + c];

    float xf[4], xr[4];
    #pragma unroll
    for (int j = 0; j < 4; ++j) {
        xf[j] = x[lane * 4 + j];
        xr[j] = x[127 - (lane * 4 + j)];
    }

    float of[4], orv[4];
    {   // forward (lr)
        float A = 1.f, S = 0.f;
        #pragma unroll
        for (int j = 0; j < 4; ++j) { S = a0 * S + b0 * xf[j]; A *= a0; }
        for (int d = 1; d < 32; d <<= 1) {
            float Ap = __shfl_up(A, d, 32);
            float Sp = __shfl_up(S, d, 32);
            if (lane >= d) { S = A * Sp + S; A = A * Ap; }
        }
        float Sx = __shfl_up(S, 1, 32);
        float s  = (lane == 0) ? 0.f : Sx;
        #pragma unroll
        for (int j = 0; j < 4; ++j) { s = a0 * s + b0 * xf[j]; of[j] = c0 * s + d0 * xf[j]; }
    }
    {   // reverse (rl), scan order t = 127-p
        float A = 1.f, S = 0.f;
        #pragma unroll
        for (int j = 0; j < 4; ++j) { S = a1 * S + b1 * xr[j]; A *= a1; }
        for (int d = 1; d < 32; d <<= 1) {
            float Ap = __shfl_up(A, d, 32);
            float Sp = __shfl_up(S, d, 32);
            if (lane >= d) { S = A * Sp + S; A = A * Ap; }
        }
        float Sx = __shfl_up(S, 1, 32);
        float s  = (lane == 0) ? 0.f : Sx;
        #pragma unroll
        for (int j = 0; j < 4; ++j) { s = a1 * s + b1 * xr[j]; orv[j] = c1 * s + d1 * xr[j]; }
    }
    // reverse result for position p=4*lane+j lives in lane 31-lane, element 3-j
    #pragma unroll
    for (int j = 0; j < 4; ++j) {
        float rv = __shfl(orv[3 - j], 31 - lane, 32);
        y[lane * 4 + j] = of[j] + rv;
    }
}

// ---------------------------------------------------------------------------
// Column scans (axis=H): tb then bt; finalize fused = 0.25*(sum4)*gate.
// One thread per (b,c,w) column -> coalesced over W.
// ---------------------------------------------------------------------------
__global__ __launch_bounds__(256)
void colscan(const float* __restrict__ X, float* __restrict__ F,
             const float* __restrict__ gate,
             const float* __restrict__ pa, const float* __restrict__ pb,
             const float* __restrict__ pc, const float* __restrict__ pd) {
    int idx = blockIdx.x * 256 + threadIdx.x;         // B*C*W threads
    int w = idx % W_;
    int c = (idx / W_) % C_;
    int b = idx / (C_ * W_);
    size_t base = ((size_t)b * C_ + c) * HW + w;
    float a2 = sigmoidf_(pa[2 * C_ + c]), b2 = pb[2 * C_ + c];
    float c2 = pc[2 * C_ + c],            d2 = pd[2 * C_ + c];
    float a3 = sigmoidf_(pa[3 * C_ + c]), b3 = pb[3 * C_ + c];
    float c3 = pc[3 * C_ + c],            d3 = pd[3 * C_ + c];
    float s = 0.f;
    for (int h = 0; h < H_; ++h) {
        size_t i = base + (size_t)h * W_;
        float xv = X[i];
        s = a2 * s + b2 * xv;
        F[i] += c2 * s + d2 * xv;
    }
    s = 0.f;
    for (int h = H_ - 1; h >= 0; --h) {
        size_t i = base + (size_t)h * W_;
        float xv = X[i];
        s = a3 * s + b3 * xv;
        float fu = 0.25f * (F[i] + c3 * s + d3 * xv);
        F[i] = fu * gate[i];
    }
}

// ---------------------------------------------------------------------------
// Host launcher
// ---------------------------------------------------------------------------
extern "C" void kernel_launch(void* const* d_in, const int* in_sizes, int n_in,
                              void* d_out, int out_size, void* d_ws, size_t ws_size,
                              hipStream_t stream) {
    (void)in_sizes; (void)n_in; (void)out_size; (void)ws_size;
    const float* x = (const float*)d_in[0];
    auto P = [&](int i) { return (const float*)d_in[i]; };

    float* ws    = (float*)d_ws;
    float* tA    = ws;                 // full tensors (B,128,HW)
    float* tB    = tA + NEL;
    float* tC    = tB + NEL;
    float* tD    = tC + NEL;           // tA..tB contiguous -> FFN hidden (2N)
    float* s1    = tD + NEL;           // bottleneck (B,16,HW)
    float* s2    = s1 + NSM;
    float* stats = s2 + NSM;           // B*G*2 floats
    __bf16* wb   = (__bf16*)(stats + 128);   // bf16 weight arena
    float* out   = (float*)d_out;

    // bf16 weight offsets
    __bf16* w_pw1[4] = { wb +     0, wb +  2048, wb +  4096, wb +  6144 };
    __bf16* w_pw2[4] = { wb +  8192, wb + 10240, wb + 12288, wb + 14336 };
    __bf16* w_in     = wb + 16384;
    __bf16* w_gate   = wb + 32768;
    __bf16* w_outpw  = wb + 49152;
    __bf16* w_ffn1   = wb + 65536;
    __bf16* w_ffn2   = wb + 98304;

    auto cv = [&](const float* s, __bf16* d, int n) {
        cvt_bf16<<<(n + 255) / 256, 256, 0, stream>>>(s, d, n);
    };
    cv(P(1),  w_pw1[0], 2048);  cv(P(7),  w_pw1[1], 2048);
    cv(P(13), w_pw1[2], 2048);  cv(P(19), w_pw1[3], 2048);
    cv(P(4),  w_pw2[0], 2048);  cv(P(10), w_pw2[1], 2048);
    cv(P(16), w_pw2[2], 2048);  cv(P(22), w_pw2[3], 2048);
    cv(P(25), w_in,    16384);  cv(P(32), w_gate,  16384);
    cv(P(35), w_outpw, 16384);
    cv(P(38), w_ffn1,  32768);  cv(P(40), w_ffn2,  32768);

    const dim3 g128(HW / 128, B_);
    const dim3 g64(HW / 64, B_);
    const int  ebF  = (int)(NEL / 256);    // full-tensor elementwise blocks
    const int  ebS  = (int)(NSM / 256);    // bottleneck elementwise blocks
    const int  gnB  = B_ * G_;

    // one GBC bottleneck block: pw1(+bias)[*mul2] -> dw -> pw2 -> GN(+silu,+resid)
    auto bott = [&](const float* in, const float* mul2, float* ob, float* fout,
                    const float* resid, int blk, int base, bool k3) {
        if (mul2)
            gemm_pw<128, 16, 128, true, false, false, false, true>
                <<<g128, 256, 0, stream>>>(in, mul2, w_pw1[blk], P(base + 1), nullptr, s1);
        else
            gemm_pw<128, 16, 128, true, false, false, false, false>
                <<<g128, 256, 0, stream>>>(in, nullptr, w_pw1[blk], P(base + 1), nullptr, s1);
        if (k3) dwconv3<CM><<<ebS, 256, 0, stream>>>(s1, P(base + 2), s2);
        else    chan_scale<CM><<<ebS, 256, 0, stream>>>(s1, P(base + 2), s2);
        gemm_pw<16, 128, 128, false, false, false, false, false>
            <<<g128, 256, 0, stream>>>(s2, nullptr, w_pw2[blk], nullptr, nullptr, ob);
        gn_stats<<<gnB, 256, 0, stream>>>(ob, stats);
        gn_apply<<<ebF, 256, 0, stream>>>(ob, P(base + 4), P(base + 5), stats,
                                          fout, 1, resid);
    };

    // ---- GBC ----
    bott(x,  nullptr, tA, tA, nullptr, 0, 1,  true);          // b1
    bott(tA, nullptr, tB, tB, nullptr, 1, 7,  true);          // b2 -> x1 = tB
    bott(x,  nullptr, tA, tA, nullptr, 2, 13, false);         // b3 -> x2 = tA
    bott(tA, tB,      tB, tA, x,       3, 19, false);         // b4(x1*x2); Y = .. + x -> tA

    // ---- SAVSS ----
    gemm_pw<128, 128, 128, false, false, false, false, false>
        <<<g128, 256, 0, stream>>>(tA, nullptr, w_in, nullptr, nullptr, tB);
    gn_stats<<<gnB, 256, 0, stream>>>(tB, stats);
    gn_apply<<<ebF, 256, 0, stream>>>(tB, P(26), P(27), stats, tB, 1, nullptr); // Xs = tB
    gemm_pw<128, 128, 128, true, false, true, false, false>
        <<<g128, 256, 0, stream>>>(tB, nullptr, w_gate, P(33), nullptr, tC);    // gate = sigmoid
    rowscan<<<(B_ * C_ * H_) / 8, 256, 0, stream>>>(
        tB, P(28), P(29), P(30), P(31), tD);                  // lr + rl -> tD
    colscan<<<(B_ * C_ * W_) / 256, 256, 0, stream>>>(
        tB, tD, tC, P(28), P(29), P(30), P(31));              // fused*gate -> tD
    dwconv3<C_><<<ebF, 256, 0, stream>>>(tD, P(34), tB);      // out_dw
    gemm_pw<128, 128, 128, false, false, false, false, false>
        <<<g128, 256, 0, stream>>>(tB, nullptr, w_outpw, nullptr, nullptr, tC); // out_pw
    gn_stats<<<gnB, 256, 0, stream>>>(tC, stats);
    gn_apply<<<ebF, 256, 0, stream>>>(tC, P(36), P(37), stats, tC, 1, tA); // Z = silu(gn)+Y

    // ---- FFN ----
    gn_stats<<<gnB, 256, 0, stream>>>(tC, stats);
    gn_apply<<<ebF, 256, 0, stream>>>(tC, P(42), P(43), stats, tD, 0, nullptr); // norm
    gemm_pw<128, 256, 64, true, true, false, false, false>
        <<<g64, 256, 0, stream>>>(tD, nullptr, w_ffn1, P(39), nullptr, tA);     // ffn1+silu
    gemm_pw<256, 128, 64, true, false, false, true, false>
        <<<g64, 256, 0, stream>>>(tA, nullptr, w_ffn2, P(41), tC, out);         // ffn2+Z
}
